// MemTransformerLM_22780506538007
// MI455X (gfx1250) — matrix operand order repaired and checked by reading, not compiled, not run on hardware
//
#include <hip/hip_runtime.h>

// ---------------------------------------------------------------------------
// Transformer-XL (6 layers) forward for MI455X / gfx1250, wave32 + WMMA bf16.
// All matmuls run on v_wmma_f32_16x16x32_bf16 with fp32 accumulation.
// GEMM: weights pre-transposed to [N,K] bf16, double-buffered LDS tiles with
// register staging so the 5 global_load_b128 per K-step issue back-to-back
// and are hidden behind the WMMA block of the previous tile.
// ---------------------------------------------------------------------------

typedef __attribute__((ext_vector_type(8)))  __bf16 v8bf;
typedef __attribute__((ext_vector_type(16))) __bf16 v16bf;
typedef __attribute__((ext_vector_type(8)))  float  v8f;

#define NL    6
#define NH    8
#define DM    512
#define DH    64
#define DI    2048
#define QLEN  256
#define BSZ   32
#define MLEN  256
#define KLEN  512   // QLEN + MLEN

// ---------------------------------------------------------------------------
// WMMA fragment helpers (gfx1250 wave32 layouts, CDNA5 ISA 7.12.2)
// ---------------------------------------------------------------------------
__device__ __forceinline__ v16bf cat8(v8bf a, v8bf b) {
  return __builtin_shufflevector(a, b, 0,1,2,3,4,5,6,7,8,9,10,11,12,13,14,15);
}

// A fragment: 16x32 (MxK), source row-major, row stride `ld` halves.
// lane L: m = L&15, hi = L>>4; elems 0..7 -> K = 8*hi + e ; 8..15 -> 16+8*hi+(e-8)
__device__ __forceinline__ v16bf load_frag_a(const __bf16* base, int ld, int lane) {
  const int m = lane & 15, hi = lane >> 4;
  const __bf16* p = base + m * ld + 8 * hi;
  v8bf lo = *(const v8bf*)p;
  v8bf hh = *(const v8bf*)(p + 16);
  return cat8(lo, hh);
}

// B fragment: 32x16 (KxN), source stored K-contiguous per column: elem [n][k]
// at base + n*ld + k.  lane L: n = L&15, hi = L>>4; elems 0..7 -> K = 8*hi + e ; 8..15 -> 16+8*hi+(e-8)
__device__ __forceinline__ v16bf load_frag_b(const __bf16* base, int ld, int lane) {
  const int n = lane & 15, hi = lane >> 4;
  const __bf16* p = base + n * ld + 8 * hi;
  v8bf lo = *(const v8bf*)p;
  v8bf hh = *(const v8bf*)(p + 16);
  return cat8(lo, hh);
}

__device__ __forceinline__ v8f wmma_bf16(v16bf a, v16bf b, v8f c) {
  return __builtin_amdgcn_wmma_f32_16x16x32_bf16(false, a, false, b,
                                                 (short)0, c, false, false);
}

// ---------------------------------------------------------------------------
// Tiled GEMM: C[M,N] (f32) = A[M,K] (bf16, row-major) @ Bt[N,K] (bf16,
// row-major = B transposed).  Block tile 64x256, BK=32, 8 waves (256 thr),
// wave tile 32x64 -> 8 WMMA per K-step; double-buffered LDS pipeline.
// ---------------------------------------------------------------------------
#define BM 64
#define BN 256
#define BK 32
#define LDA 40   // halves; 80B rows keep v8bf loads 16B aligned
#define LDB 40

__global__ __launch_bounds__(256) void k_gemm_bf16(
    const __bf16* __restrict__ A, const __bf16* __restrict__ Bt,
    float* __restrict__ C, int M, int N, int K) {
  __shared__ __bf16 sa[2][BM * LDA];   // 2 x  5.1 KB
  __shared__ __bf16 sb[2][BN * LDB];   // 2 x 20.5 KB

  const int t    = threadIdx.x;
  const int lane = t & 31;
  const int wave = t >> 5;
  const int wm   = wave & 1;        // 2 waves along M (2*32 = 64)
  const int wn   = wave >> 1;       // 4 waves along N (4*64 = 256)
  const int mb   = blockIdx.y * BM;
  const int nb   = blockIdx.x * BN;

  // loader mapping: A -> thread t copies 8 halves of row (t>>2);
  //                 B -> thread t copies the 32-half row (nb+t).
  const int arow = t >> 2;
  const int aseg = (t & 3) * 8;
  const __bf16* aptr = A + (size_t)(mb + arow) * K + aseg;
  const __bf16* bptr = Bt + (size_t)(nb + t) * K;

  const v8f z = {0.f,0.f,0.f,0.f,0.f,0.f,0.f,0.f};
  v8f acc0[4], acc1[4];
  #pragma unroll
  for (int s = 0; s < 4; ++s) { acc0[s] = z; acc1[s] = z; }

  const int nt = K / BK;

  // prologue: stage tile 0
  v8bf areg = *(const v8bf*)aptr;
  v8bf breg[4];
  #pragma unroll
  for (int q = 0; q < 4; ++q) breg[q] = *(const v8bf*)(bptr + q * 8);
  *(v8bf*)(&sa[0][arow * LDA + aseg]) = areg;
  #pragma unroll
  for (int q = 0; q < 4; ++q) *(v8bf*)(&sb[0][t * LDB + q * 8]) = breg[q];

  for (int i = 0; i < nt; ++i) {
    __syncthreads();                       // buf[i&1] ready for everyone
    const int cur  = i & 1;
    const bool more = (i + 1 < nt);
    if (more) {                            // issue next tile's global loads now
      const int k = (i + 1) * BK;
      areg = *(const v8bf*)(aptr + k);
      #pragma unroll
      for (int q = 0; q < 4; ++q) breg[q] = *(const v8bf*)(bptr + k + q * 8);
      if (i + 2 < nt) {                    // prefetch tile i+2 into L2
        __builtin_prefetch(aptr + k + BK, 0, 1);
        __builtin_prefetch(bptr + k + BK, 0, 1);
      }
    }
    // compute tile i from LDS while the loads above are in flight
    v16bf a0 = load_frag_a(&sa[cur][(wm * 32) * LDA],      LDA, lane);
    v16bf a1 = load_frag_a(&sa[cur][(wm * 32 + 16) * LDA], LDA, lane);
    #pragma unroll
    for (int s = 0; s < 4; ++s) {
      v16bf bfR = load_frag_b(&sb[cur][(wn * 64 + s * 16) * LDB], LDB, lane);
      acc0[s] = wmma_bf16(a0, bfR, acc0[s]);
      acc1[s] = wmma_bf16(a1, bfR, acc1[s]);
    }
    if (more) {
      __syncthreads();                     // everyone done reading buf[cur^1]
      *(v8bf*)(&sa[cur ^ 1][arow * LDA + aseg]) = areg;
      #pragma unroll
      for (int q = 0; q < 4; ++q)
        *(v8bf*)(&sb[cur ^ 1][t * LDB + q * 8]) = breg[q];
    }
  }

  const int cn = lane & 15, chi = lane >> 4;
  #pragma unroll
  for (int s = 0; s < 4; ++s) {
    const int col = nb + wn * 64 + s * 16 + cn;
    #pragma unroll
    for (int v = 0; v < 8; ++v) {
      const int r0 = mb + wm * 32 + v + 8 * chi;
      C[(size_t)r0 * N + col]        = acc0[s][v];
      C[(size_t)(r0 + 16) * N + col] = acc1[s][v];
    }
  }
}

// ---------------------------------------------------------------------------
// Fused relative attention.  One wave per (b, head, 16-query-row tile).
// Dynamic LDS: bd[16][512] f32 | sS[16][512] f32 | p[16][512] bf16 = 80 KB.
// ---------------------------------------------------------------------------
__global__ __launch_bounds__(32) void k_attn(
    const __bf16* __restrict__ qw,  // [b,n,256,64] q + r_w_bias
    const __bf16* __restrict__ qr,  // [b,n,256,64] q + r_r_bias
    const __bf16* __restrict__ kb,  // [b,n,512,64]
    const __bf16* __restrict__ vt,  // [b,n,64,512] (V transposed)
    const __bf16* __restrict__ rk,  // [n,512,64]   r_head_k
    __bf16* __restrict__ vec) {     // [i,b,512]
  extern __shared__ char smem[];
  float*  bd = (float*)smem;                 // raw BD (pre rel-shift)
  float*  sS = bd + 16 * KLEN;               // scores
  __bf16* pp = (__bf16*)(sS + 16 * KLEN);    // softmax probs (bf16)

  const int it   = blockIdx.x;               // q tile (16 rows)
  const int n    = blockIdx.y;
  const int b    = blockIdx.z;
  const int i0   = it * 16;
  const int lane = threadIdx.x;
  const int cn   = lane & 15, chi = lane >> 4;

  const __bf16* qw_p = qw + ((size_t)(b * NH + n) * QLEN + i0) * DH;
  const __bf16* qr_p = qr + ((size_t)(b * NH + n) * QLEN + i0) * DH;
  const __bf16* k_p  = kb + (size_t)(b * NH + n) * KLEN * DH;
  const __bf16* vt_p = vt + (size_t)(b * NH + n) * DH * KLEN;
  const __bf16* rk_p = rk + (size_t)n * KLEN * DH;

  const v8f z = {0.f,0.f,0.f,0.f,0.f,0.f,0.f,0.f};

  v16bf aqr0 = load_frag_a(qr_p,      DH, lane);
  v16bf aqr1 = load_frag_a(qr_p + 32, DH, lane);
  v16bf aqw0 = load_frag_a(qw_p,      DH, lane);
  v16bf aqw1 = load_frag_a(qw_p + 32, DH, lane);

  // Phase 1: raw BD[i, jj] = (q_i + r_r_bias) . r_head_k[jj]
  for (int jb = 0; jb < KLEN / 16; ++jb) {
    v8f a = z;
    a = wmma_bf16(aqr0, load_frag_b(rk_p + jb * 16 * DH,      DH, lane), a);
    a = wmma_bf16(aqr1, load_frag_b(rk_p + jb * 16 * DH + 32, DH, lane), a);
    #pragma unroll
    for (int v = 0; v < 8; ++v)
      bd[(v + 8 * chi) * KLEN + jb * 16 + cn] = a[v];
  }

  // Phase 2: AC + rel-shifted BD, scale, causal+mem mask
  for (int jb = 0; jb < KLEN / 16; ++jb) {
    v8f a = z;
    a = wmma_bf16(aqw0, load_frag_b(k_p + jb * 16 * DH,      DH, lane), a);
    a = wmma_bf16(aqw1, load_frag_b(k_p + jb * 16 * DH + 32, DH, lane), a);
    const int j = jb * 16 + cn;
    #pragma unroll
    for (int v = 0; v < 8; ++v) {
      const int il = v + 8 * chi;
      const int i  = i0 + il;
      const int jj = j + (QLEN - 1) - i;     // rel_shift: BD[i,j]=Braw[i,jj]
      float s;
      if (j > i + MLEN) s = -3.0e38f;        // masked (jj also out of range)
      else              s = (a[v] + bd[il * KLEN + jj]) * 0.125f;  // 1/sqrt(64)
      sS[il * KLEN + j] = s;
    }
  }

  // Phase 3: softmax over j (512) per row, wave32 shuffle reductions
  for (int r = 0; r < 16; ++r) {
    float* row = sS + r * KLEN;
    const int cb = lane * 16;
    float vals[16];
    float m = -3.4e38f;
    #pragma unroll
    for (int q = 0; q < 16; ++q) { vals[q] = row[cb + q]; m = fmaxf(m, vals[q]); }
    #pragma unroll
    for (int off = 16; off > 0; off >>= 1) m = fmaxf(m, __shfl_xor(m, off));
    float sum = 0.f;
    #pragma unroll
    for (int q = 0; q < 16; ++q) { float e = __expf(vals[q] - m); vals[q] = e; sum += e; }
    #pragma unroll
    for (int off = 16; off > 0; off >>= 1) sum += __shfl_xor(sum, off);
    const float inv = 1.0f / sum;
    #pragma unroll
    for (int q = 0; q < 16; ++q) pp[r * KLEN + cb + q] = (__bf16)(vals[q] * inv);
  }

  // Phase 4: O[16,64] = P[16,512] @ V[512,64]
  v8f o[4];
  #pragma unroll
  for (int s = 0; s < 4; ++s) o[s] = z;
  for (int kc = 0; kc < KLEN / 32; ++kc) {
    v16bf af = load_frag_a(pp + kc * 32, KLEN, lane);
    #pragma unroll
    for (int s = 0; s < 4; ++s)
      o[s] = wmma_bf16(af, load_frag_b(vt_p + (s * 16) * KLEN + kc * 32, KLEN, lane), o[s]);
  }
  #pragma unroll
  for (int s = 0; s < 4; ++s) {
    const int d = n * DH + s * 16 + cn;
    #pragma unroll
    for (int v = 0; v < 8; ++v) {
      const int i = i0 + v + 8 * chi;
      vec[((size_t)i * BSZ + b) * (NH * DH) + d] = (__bf16)o[s][v];
    }
  }
}

// ---------------------------------------------------------------------------
// Small support kernels (2D grids -> no integer division per thread)
// ---------------------------------------------------------------------------
__global__ void k_copy_f32(const float* __restrict__ in, float* __restrict__ out, int n) {
  int i = blockIdx.x * 256 + threadIdx.x;
  if (i < n) out[i] = in[i];
}

// weight convert + transpose: in f32 [K,N] row-major -> out bf16 [N,K]
// grid: (N/256, K)
__global__ void k_w_to_bf16_t(const float* __restrict__ in, __bf16* __restrict__ out,
                              int K, int N) {
  const int n = blockIdx.x * 256 + threadIdx.x;
  const int k = blockIdx.y;
  out[(size_t)n * K + k] = (__bf16)in[(size_t)k * N + n];
}

// sinusoidal relative positional embedding -> bf16 [512,512]; grid (2, 512)
__global__ void k_pos(__bf16* __restrict__ rbf) {
  const int c = blockIdx.x * 256 + threadIdx.x;
  const int j = blockIdx.y;
  float pos = (float)(KLEN - 1 - j);
  float out;
  if (c < 256) out = __sinf(pos * __powf(10000.0f, -(float)c / 256.0f));
  else         out = __cosf(pos * __powf(10000.0f, -(float)(c - 256) / 256.0f));
  rbf[(size_t)j * DM + c] = (__bf16)out;
}

// cat = concat(mems[l], h) as bf16, rows = (j, b); grid (BSZ*DM/256, KLEN)
__global__ void k_concat(const float* __restrict__ mems_l, const float* __restrict__ h,
                         __bf16* __restrict__ cat) {
  const int rb = blockIdx.x * 256 + threadIdx.x;
  const int j  = blockIdx.y;
  float v = (j < MLEN) ? mems_l[(size_t)j * BSZ * DM + rb]
                       : h[(size_t)(j - MLEN) * BSZ * DM + rb];
  cat[(size_t)j * BSZ * DM + rb] = (__bf16)v;
}

// scatter w_heads [16384,1536] f32 -> qw/qr (bias-added), K, V^T (bf16)
// grid: (1536/256, 16384)
__global__ void k_epi_qkv(const float* __restrict__ scr,
                          const float* __restrict__ rwb, const float* __restrict__ rrb,
                          __bf16* __restrict__ qw, __bf16* __restrict__ qr,
                          __bf16* __restrict__ kb, __bf16* __restrict__ vt) {
  const int c   = blockIdx.x * 256 + threadIdx.x;   // 0..1535
  const int row = blockIdx.y;                       // 0..16383
  const int j = row >> 5, b = row & 31;
  float v = scr[(size_t)row * (3 * NH * DH) + c];
  if (c < NH * DH) {                     // q: only last QLEN rows
    if (j >= MLEN) {
      int i = j - MLEN, n = c >> 6, d = c & 63;
      size_t o = ((size_t)(b * NH + n) * QLEN + i) * DH + d;
      qw[o] = (__bf16)(v + rwb[n * DH + d]);
      qr[o] = (__bf16)(v + rrb[n * DH + d]);
    }
  } else if (c < 2 * NH * DH) {          // k
    int cc = c - NH * DH, n = cc >> 6, d = cc & 63;
    kb[((size_t)(b * NH + n) * KLEN + j) * DH + d] = (__bf16)v;
  } else {                               // v, stored transposed [d][j]
    int cc = c - 2 * NH * DH, n = cc >> 6, d = cc & 63;
    vt[((size_t)(b * NH + n) * DH + d) * KLEN + j] = (__bf16)v;
  }
}

// r @ r_net_w result [512,512] f32 -> rk bf16 [n][j][d]; grid (2, 512)
__global__ void k_epi_rk(const float* __restrict__ scr2, __bf16* __restrict__ rk) {
  const int c = blockIdx.x * 256 + threadIdx.x;
  const int j = blockIdx.y;
  const int n = c >> 6, d = c & 63;
  rk[((size_t)n * KLEN + j) * DH + d] = (__bf16)scr2[(size_t)j * DM + c];
}

// x = resid + y (+ bias); LayerNorm -> h (f32) and h (bf16). One wave per row.
__global__ __launch_bounds__(256) void k_add_ln(
    const float* resid, const float* y, const float* bias, int use_bias,
    const float* __restrict__ g, const float* __restrict__ be,
    float* hout, __bf16* __restrict__ hbf) {
  const int row  = blockIdx.x * 8 + (threadIdx.x >> 5);
  const int lane = threadIdx.x & 31;
  const float* r0 = resid + (size_t)row * DM;
  const float* y0 = y + (size_t)row * DM;
  float vals[16];
  float sum = 0.f, sq = 0.f;
  #pragma unroll
  for (int q = 0; q < 16; ++q) {
    int c = q * 32 + lane;
    float x = r0[c] + y0[c] + (use_bias ? bias[c] : 0.0f);
    vals[q] = x; sum += x; sq += x * x;
  }
  #pragma unroll
  for (int off = 16; off > 0; off >>= 1) {
    sum += __shfl_xor(sum, off);
    sq  += __shfl_xor(sq, off);
  }
  const float mean = sum * (1.0f / DM);
  const float var  = sq * (1.0f / DM) - mean * mean;
  const float rstd = rsqrtf(var + 1e-5f);
  #pragma unroll
  for (int q = 0; q < 16; ++q) {
    int c = q * 32 + lane;
    float o = (vals[q] - mean) * rstd * g[c] + be[c];
    hout[(size_t)row * DM + c] = o;
    hbf[(size_t)row * DM + c]  = (__bf16)o;
  }
}

// ff1 = bf16(relu(scr + b1)); grid (DI/256, 8192)
__global__ void k_bias_relu(const float* __restrict__ scr, const float* __restrict__ b1,
                            __bf16* __restrict__ out) {
  const int c   = blockIdx.x * 256 + threadIdx.x;
  const int row = blockIdx.y;
  const size_t idx = (size_t)row * DI + c;
  float v = scr[idx] + b1[c];
  out[idx] = (__bf16)fmaxf(v, 0.0f);
}

// ---------------------------------------------------------------------------
// Host orchestration
// ---------------------------------------------------------------------------
static inline dim3 grid1(size_t n) { return dim3((unsigned)((n + 255) / 256)); }

extern "C" void kernel_launch(void* const* d_in, const int* in_sizes, int n_in,
                              void* d_out, int out_size, void* d_ws, size_t ws_size,
                              hipStream_t stream) {
  const float* w      = (const float*)d_in[0];
  const float* mems   = (const float*)d_in[1];
  const float* qkv_w  = (const float*)d_in[2];
  const float* rnet_w = (const float*)d_in[3];
  const float* o_w    = (const float*)d_in[4];
  const float* ln1_g  = (const float*)d_in[5];
  const float* ln1_b  = (const float*)d_in[6];
  const float* ff_w1  = (const float*)d_in[7];
  const float* ff_b1  = (const float*)d_in[8];
  const float* ff_w2  = (const float*)d_in[9];
  const float* ff_b2  = (const float*)d_in[10];
  const float* ln2_g  = (const float*)d_in[11];
  const float* ln2_b  = (const float*)d_in[12];
  const float* rwb    = (const float*)d_in[13];
  const float* rrb    = (const float*)d_in[14];

  // workspace layout (all chunks are 256B aligned by construction)
  char* ws = (char*)d_ws;
  size_t off = 0;
  auto take = [&](size_t bytes) { char* p = ws + off; off += bytes; return p; };
  float*  h32  = (float*)take((size_t)QLEN * BSZ * DM * 4);          // 16 MB
  __bf16* cat  = (__bf16*)take((size_t)KLEN * BSZ * DM * 2);         // 16 MB
  __bf16* wqkv = (__bf16*)take((size_t)DM * 3 * NH * DH * 2);        // 1.5 MB  [N,K]
  __bf16* wrn  = (__bf16*)take((size_t)DM * NH * DH * 2);            // 0.5 MB  [N,K]
  __bf16* wo   = (__bf16*)take((size_t)NH * DH * DM * 2);            // 0.5 MB  [N,K]
  __bf16* wf1  = (__bf16*)take((size_t)DM * DI * 2);                 // 2 MB    [N,K]
  __bf16* wf2  = (__bf16*)take((size_t)DI * DM * 2);                 // 2 MB    [N,K]
  __bf16* rbf  = (__bf16*)take((size_t)KLEN * DM * 2);               // 0.5 MB
  __bf16* rk   = (__bf16*)take((size_t)NH * KLEN * DH * 2);          // 0.5 MB
  __bf16* qw   = (__bf16*)take((size_t)BSZ * NH * QLEN * DH * 2);    // 8 MB
  __bf16* qr   = (__bf16*)take((size_t)BSZ * NH * QLEN * DH * 2);    // 8 MB
  __bf16* kb   = (__bf16*)take((size_t)BSZ * NH * KLEN * DH * 2);    // 16 MB
  __bf16* vt   = (__bf16*)take((size_t)BSZ * NH * DH * KLEN * 2);    // 16 MB
  __bf16* vec  = (__bf16*)take((size_t)QLEN * BSZ * NH * DH * 2);    // 8 MB
  __bf16* hbf  = (__bf16*)take((size_t)QLEN * BSZ * DM * 2);         // 8 MB
  __bf16* ff1  = (__bf16*)take((size_t)QLEN * BSZ * DI * 2);         // 32 MB
  float*  scr2 = (float*)take((size_t)KLEN * NH * DH * 4);           // 1 MB
  float*  scr  = (float*)take((size_t)KLEN * BSZ * 3 * NH * DH * 4); // 96 MB
  (void)ws_size; (void)n_in; (void)in_sizes; (void)out_size;

  const size_t MQKV = (size_t)KLEN * BSZ;        // 16384
  const size_t MH   = (size_t)QLEN * BSZ;        // 8192

  // h0 = w ; positional embedding (layer-invariant)
  k_copy_f32<<<grid1(MH * DM), 256, 0, stream>>>(w, h32, (int)(MH * DM));
  k_pos<<<dim3(DM / 256, KLEN), 256, 0, stream>>>(rbf);

  for (int l = 0; l < NL; ++l) {
    // weights -> bf16, transposed to [N,K]
    k_w_to_bf16_t<<<dim3(3 * NH * DH / 256, DM), 256, 0, stream>>>(
        qkv_w + (size_t)l * DM * 3 * NH * DH, wqkv, DM, 3 * NH * DH);
    k_w_to_bf16_t<<<dim3(NH * DH / 256, DM), 256, 0, stream>>>(
        rnet_w + (size_t)l * DM * NH * DH, wrn, DM, NH * DH);
    k_w_to_bf16_t<<<dim3(DM / 256, NH * DH), 256, 0, stream>>>(
        o_w + (size_t)l * NH * DH * DM, wo, NH * DH, DM);
    k_w_to_bf16_t<<<dim3(DI / 256, DM), 256, 0, stream>>>(
        ff_w1 + (size_t)l * DM * DI, wf1, DM, DI);
    k_w_to_bf16_t<<<dim3(DM / 256, DI), 256, 0, stream>>>(
        ff_w2 + (size_t)l * DI * DM, wf2, DI, DM);

    // cat = [mems_l ; h]
    k_concat<<<dim3(BSZ * DM / 256, KLEN), 256, 0, stream>>>(
        mems + (size_t)l * MLEN * BSZ * DM, h32, cat);

    // QKV projection: [16384,512] @ [512,1536]
    k_gemm_bf16<<<dim3(3 * NH * DH / BN, MQKV / BM), 256, 0, stream>>>(
        cat, wqkv, scr, (int)MQKV, 3 * NH * DH, DM);
    k_epi_qkv<<<dim3(3 * NH * DH / 256, (unsigned)MQKV), 256, 0, stream>>>(
        scr, rwb, rrb, qw, qr, kb, vt);

    // r_head_k = r @ r_net_w : [512,512] @ [512,512]
    k_gemm_bf16<<<dim3(NH * DH / BN, KLEN / BM), 256, 0, stream>>>(
        rbf, wrn, scr2, KLEN, NH * DH, DM);
    k_epi_rk<<<dim3(NH * DH / 256, KLEN), 256, 0, stream>>>(scr2, rk);

    // fused relative attention (AC + rel-shifted BD + softmax + PV)
    k_attn<<<dim3(QLEN / 16, NH, BSZ), 32,
             (16 * KLEN * 4) * 2 + 16 * KLEN * 2, stream>>>(
        qw, qr, kb, vt, rk, vec);

    // output projection + residual + LN1
    k_gemm_bf16<<<dim3(DM / BN, MH / BM), 256, 0, stream>>>(
        vec, wo, scr, (int)MH, DM, NH * DH);
    k_add_ln<<<dim3((unsigned)(MH / 8)), 256, 0, stream>>>(
        h32, scr, nullptr, 0, ln1_g + (size_t)l * DM, ln1_b + (size_t)l * DM,
        h32, hbf);

    // FFN
    k_gemm_bf16<<<dim3(DI / BN, MH / BM), 256, 0, stream>>>(
        hbf, wf1, scr, (int)MH, DI, DM);
    k_bias_relu<<<dim3(DI / 256, (unsigned)MH), 256, 0, stream>>>(
        scr, ff_b1 + (size_t)l * DI, ff1);
    k_gemm_bf16<<<dim3(DM / BN, MH / BM), 256, 0, stream>>>(
        ff1, wf2, scr, (int)MH, DM, DI);
    k_add_ln<<<dim3((unsigned)(MH / 8)), 256, 0, stream>>>(
        h32, scr, ff_b2 + (size_t)l * DM, 1,
        ln2_g + (size_t)l * DM, ln2_b + (size_t)l * DM, h32, hbf);
  }

  k_copy_f32<<<grid1(MH * DM), 256, 0, stream>>>(h32, (float*)d_out, (int)(MH * DM));
}
